// Embedding_32813550141833
// MI455X (gfx1250) — matrix operand order repaired and checked by reading, not compiled, not run on hardware
//
#include <hip/hip_runtime.h>

// Problem constants (from reference)
#define E_   256
#define F_   32
#define P_   4
#define U_   10000
#define I_   10000
#define B_   512
#define HW_  64                 // (F/P)^2
#define TOK_ 65                 // HW_ + 1 (cls)

#define ECHUNK   8              // e values per K-chunk
#define KCHUNK   128            // ECHUNK * 16 (p*4+q)
#define LDROW    144            // f16 per LDS row: 128 data + 16 pad (288 B, 16B-aligned rows)
#define WROW     (E_ * P_ * P_) // 4096 f16 per output-channel row of converted weights

typedef __attribute__((ext_vector_type(16))) _Float16 v16h;
typedef __attribute__((ext_vector_type(8)))  _Float16 v8h;
typedef __attribute__((ext_vector_type(8)))  float    v8f;
typedef __attribute__((ext_vector_type(4)))  float    v4f;

static __device__ __forceinline__ v16h cat8(v8h lo, v8h hi) {
    return __builtin_shufflevector(lo, hi, 0,1,2,3,4,5,6,7,8,9,10,11,12,13,14,15);
}

// ---------------------------------------------------------------------------
// Prep: convert conv_w (256x256x4x4 f32) to f16 once into workspace.
// Layout preserved: w16[o][e*16 + p*4 + q], 4096 f16 per o (8 KB rows).
// ---------------------------------------------------------------------------
__global__ __launch_bounds__(256)
void convert_w_to_f16(const float* __restrict__ w, _Float16* __restrict__ w16) {
    const int i = (blockIdx.x * 256 + threadIdx.x) * 4;
    v4f v = *(const v4f*)(w + i);
    w16[i + 0] = (_Float16)v.x;
    w16[i + 1] = (_Float16)v.y;
    w16[i + 2] = (_Float16)v.z;
    w16[i + 3] = (_Float16)v.w;
}

// ---------------------------------------------------------------------------
// Main: one workgroup (256 threads = 8 wave32) per batch element b.
// Wave w computes y tile: all 64 hw rows x output channels [32w, 32w+32).
// B tile (f16 weights) is streamed global->LDS with async-to-LDS copies,
// overlapped with the in-LDS outer-product build of the A tile.
// ---------------------------------------------------------------------------
__global__ __launch_bounds__(256)
void fused_embed_outer_conv(const int*      __restrict__ user,
                            const int*      __restrict__ item,
                            const float*    __restrict__ user_tables,
                            const float*    __restrict__ item_tables,
                            const _Float16* __restrict__ w16,     // converted conv_w
                            const float*    __restrict__ conv_b,
                            const float*    __restrict__ cls_token,
                            const float*    __restrict__ positions,
                            float* __restrict__ out_x,    // (B, 65, 256)
                            float* __restrict__ out_eu,   // (B, 256, 32)
                            float* __restrict__ out_ei)   // (B, 256, 32)
{
    __shared__ __attribute__((aligned(16))) float    u_s [ECHUNK][F_];
    __shared__ __attribute__((aligned(16))) float    it_s[ECHUNK][F_];
    __shared__ __attribute__((aligned(16))) _Float16 A_s [HW_][LDROW];   // A[s][k]  (f16)
    __shared__ __attribute__((aligned(16))) _Float16 Bt_s[E_ ][LDROW];   // B^T[o][k] (f16)

    const int b    = blockIdx.x;
    const int tid  = threadIdx.x;
    const int lane = tid & 31;            // wave32
    const int wave = tid >> 5;            // 0..7
    const int nl   = lane & 15;           // n (or m) within 16x16 tile
    const int half = lane >> 4;           // lane half selects K sub-range

    const int uidx = user[b];
    const int iidx = item[b];

    // cls row: x[b,0,o] = cls_token[o] + positions[0,o]
    out_x[(size_t)b * TOK_ * E_ + tid] = cls_token[tid] + positions[tid];

    v8f acc[4][2];
#pragma unroll
    for (int mt = 0; mt < 4; ++mt)
#pragma unroll
        for (int nt = 0; nt < 2; ++nt)
            acc[mt][nt] = (v8f)0.0f;

    const int n_base = wave * 32;
    // LDS byte address of this thread's B^T row (generic addr low 32 bits = LDS offset)
    const unsigned lds_brow = (unsigned)(uintptr_t)&Bt_s[tid][0];

    for (int e0 = 0; e0 < E_; e0 += ECHUNK) {
        __syncthreads();   // previous iteration's WMMA reads of A_s/Bt_s done

        // ---- async-copy B^T tile: thread t owns o = t; 256 contiguous bytes of f16 ----
        {
            const unsigned goff = (unsigned)tid * (WROW * 2) + (unsigned)e0 * 32; // bytes
            // GVS mode: offset applies to BOTH lds (vdst) and global (saddr+vaddr) sides.
#define ASYNC_B128(OFF) \
            asm volatile("global_load_async_to_lds_b128 %0, %1, %2 offset:" #OFF \
                         :: "v"(lds_brow), "v"(goff), "s"(w16) : "memory")
            ASYNC_B128(0);   ASYNC_B128(16);  ASYNC_B128(32);  ASYNC_B128(48);
            ASYNC_B128(64);  ASYNC_B128(80);  ASYNC_B128(96);  ASYNC_B128(112);
            ASYNC_B128(128); ASYNC_B128(144); ASYNC_B128(160); ASYNC_B128(176);
            ASYNC_B128(192); ASYNC_B128(208); ASYNC_B128(224); ASYNC_B128(240);
#undef ASYNC_B128
        }

        // ---- stage u/it rows for this e-chunk; also emit embed outputs ----
        {
            const int e_l = tid >> 5;     // 0..7
            const int f   = tid & 31;     // 0..31
            const float uv = user_tables[(size_t)(e0 + e_l) * (U_ * F_) + (size_t)uidx * F_ + f];
            const float iv = item_tables[(size_t)(e0 + e_l) * (I_ * F_) + (size_t)iidx * F_ + f];
            u_s [e_l][f] = uv;
            it_s[e_l][f] = iv;
            out_eu[(size_t)b * E_ * F_ + (size_t)(e0 + e_l) * F_ + f] = uv;
            out_ei[(size_t)b * E_ * F_ + (size_t)(e0 + e_l) * F_ + f] = iv;
        }
        __syncthreads();   // u_s/it_s ready for A build (asyncs still in flight)

        // ---- build A tile (overlaps async B copy): A[s][k] = u[4h+p] * it[4w+q] ----
        {
            const int s  = tid >> 2;            // 0..63
            const int kb = (tid & 3) * 32;      // this thread covers k = kb..kb+31
            const int h  = s >> 3, w = s & 7;
            _Float16* arow = &A_s[s][0];
#pragma unroll
            for (int kk = 0; kk < 32; ++kk) {
                const int k   = kb + kk;
                const int e_l = k >> 4;
                const int pq  = k & 15;
                const float v = u_s[e_l][4 * h + (pq >> 2)] * it_s[e_l][4 * w + (pq & 3)];
                arow[k] = (_Float16)v;
            }
        }

        // all async B beats landed in LDS, then rendezvous
        asm volatile("s_wait_asynccnt 0x0" ::: "memory");
        __syncthreads();   // A_s/Bt_s ready for WMMA

        // ---- 4 K-substeps of v_wmma_f32_16x16x32_f16 ----
#pragma unroll
        for (int ks = 0; ks < 4; ++ks) {
            const int kbase = ks * 32;

            v16h afrag[4];
#pragma unroll
            for (int mt = 0; mt < 4; ++mt) {
                // A layout (ISA 7.12.2): lane half 0: K[0..8)+[16..24); half 1: K[8..16)+[24..32)
                const _Float16* row = &A_s[mt * 16 + nl][0];
                v8h lo = *(const v8h*)(row + kbase + half * 8);
                v8h hi = *(const v8h*)(row + kbase + 16 + half * 8);
                afrag[mt] = cat8(lo, hi);
            }
            v16h bfrag[2];
#pragma unroll
            for (int nt = 0; nt < 2; ++nt) {
                // B layout: fixed n = lane&15; lane half 0: K[0..8)+[16..24), half 1: K[8..16)+[24..32)
                const _Float16* row = &Bt_s[n_base + nt * 16 + nl][0];
                v8h lo = *(const v8h*)(row + kbase + half * 8);
                v8h hi = *(const v8h*)(row + kbase + 16 + half * 8);
                bfrag[nt] = cat8(lo, hi);
            }
#pragma unroll
            for (int mt = 0; mt < 4; ++mt)
#pragma unroll
                for (int nt = 0; nt < 2; ++nt)
                    acc[mt][nt] = __builtin_amdgcn_wmma_f32_16x16x32_f16(
                        /*neg_a=*/false, afrag[mt],
                        /*neg_b=*/false, bfrag[nt],
                        /*c_mod=*/(short)0, acc[mt][nt],
                        /*reuse_a=*/false, /*reuse_b=*/false);
        }
    }

    // ---- epilogue: x[b, 1+s, o] = y[o,s] + conv_b[o] + positions[1+s, o] ----
    float* xrow = out_x + (size_t)b * TOK_ * E_ + E_;   // skip cls row
#pragma unroll
    for (int nt = 0; nt < 2; ++nt) {
        const int o  = n_base + nt * 16 + nl;
        const float cb = conv_b[o];
#pragma unroll
        for (int mt = 0; mt < 4; ++mt) {
#pragma unroll
            for (int r = 0; r < 8; ++r) {
                // C/D layout: VGPR r holds M = r (lanes 0-15) / r+8 (lanes 16-31)
                const int s = mt * 16 + r + half * 8;
                const float v = acc[mt][nt][r] + cb + positions[(size_t)(1 + s) * E_ + o];
                xrow[(size_t)s * E_ + o] = v;
            }
        }
    }
}

extern "C" void kernel_launch(void* const* d_in, const int* in_sizes, int n_in,
                              void* d_out, int out_size, void* d_ws, size_t ws_size,
                              hipStream_t stream) {
    const int*   user = (const int*)d_in[0];
    const int*   item = (const int*)d_in[1];
    const float* ut   = (const float*)d_in[2];
    const float* itb  = (const float*)d_in[3];
    const float* cw   = (const float*)d_in[4];
    const float* cb   = (const float*)d_in[5];
    const float* cls  = (const float*)d_in[6];
    const float* pos  = (const float*)d_in[7];

    float* out_x  = (float*)d_out;                         // B*65*256
    float* out_eu = out_x  + (size_t)B_ * TOK_ * E_;       // B*256*32
    float* out_ei = out_eu + (size_t)B_ * E_ * F_;         // B*256*32

    _Float16* w16 = (_Float16*)d_ws;                       // 256*4096 f16 = 2 MB scratch

    // 1) one-shot weight conversion (2 MB f16, stays hot in 192 MB L2)
    hipLaunchKernelGGL(convert_w_to_f16,
                       dim3((E_ * WROW) / (256 * 4)), dim3(256), 0, stream, cw, w16);

    // 2) fused gather + outer-product + conv-as-GEMM + epilogue
    hipLaunchKernelGGL(fused_embed_outer_conv,
                       dim3(B_), dim3(256), 0, stream,
                       user, item, ut, itb, w16, cb, cls, pos,
                       out_x, out_eu, out_ei);
}
